// BBADJknet_20667382628944
// MI455X (gfx1250) — matrix operand order repaired and checked by reading, not compiled, not run on hardware
//
#include <hip/hip_runtime.h>
#include <hip/hip_bf16.h>

// ---------------------------------------------------------------------------
// Types for CDNA5 WMMA (gfx1250, wave32)
// ---------------------------------------------------------------------------
typedef __attribute__((ext_vector_type(16))) __bf16    v16bf;
typedef __attribute__((ext_vector_type(8)))  float     v8f;
typedef __attribute__((ext_vector_type(8)))  unsigned  v8u;

// Round-to-nearest-even fp32 -> bf16
__device__ __forceinline__ unsigned short f2bf(float f) {
    unsigned u = __float_as_uint(f);
    u += 0x7FFFu + ((u >> 16) & 1u);
    return (unsigned short)(u >> 16);
}

// Two contiguous-ish b128 loads -> 8 dwords (16 bf16). `second` = uint4 index
// of the second 16B run relative to p (A: 2 -> +32B gap pattern; B: 1 -> dense).
__device__ __forceinline__ v8u ld8(const uint4* __restrict__ p, int second) {
    uint4 lo = p[0], hi = p[second];
    v8u r;
    r[0] = lo.x; r[1] = lo.y; r[2] = lo.z; r[3] = lo.w;
    r[4] = hi.x; r[5] = hi.y; r[6] = hi.z; r[7] = hi.w;
    return r;
}

// ---------------------------------------------------------------------------
// Pack a [K,64] fp32 weight into FRAGMENT-MAJOR bf16 pair-dwords so that each
// lane's B fragment is 8 consecutive dwords (-> two b128 loads):
//   out[(((c*64 + n)*2 + khalf)*8 + v] = { W[k,n], W[k+1,n] },
//   k = c*32 + 8*khalf + 2*(v&3) + 16*(v>>2)   (c = K/32 chunk, n = 0..63 column)
// ---------------------------------------------------------------------------
__global__ void pack_w_kernel(const float* __restrict__ W, unsigned* __restrict__ out, int K) {
    int t = blockIdx.x * blockDim.x + threadIdx.x;
    int total = (K / 32) * 1024;            // (K/32) chunks * 64 n * 2 khalf * 8 v
    if (t >= total) return;
    int v     = t & 7;
    int khalf = (t >> 3) & 1;
    int n     = (t >> 4) & 63;
    int c     = t >> 10;
    int k     = c * 32 + 8 * khalf + 2 * (v & 3) + 16 * (v >> 2);
    unsigned short lo = f2bf(W[(size_t)k * 64 + n]);
    unsigned short hi = f2bf(W[(size_t)(k + 1) * 64 + n]);
    out[t] = (unsigned)lo | ((unsigned)hi << 16);
}

// fp32 -> bf16 bulk convert, optional per-row scale (deferred deg_inv)
__global__ void cvt_bf16_kernel(const float* __restrict__ in,
                                const float* __restrict__ rowscale, int d,
                                unsigned short* __restrict__ out, size_t count) {
    size_t t = (size_t)blockIdx.x * blockDim.x + threadIdx.x;
    if (t >= count) return;
    float v = in[t];
    if (rowscale) v *= rowscale[t / d];
    out[t] = f2bf(v);
}

// ---------------------------------------------------------------------------
// Degree: deg[dst] += 1 ; then deg = 1/max(deg,1) in place
// ---------------------------------------------------------------------------
__global__ void deg_kernel(const int* __restrict__ dst, float* __restrict__ deg, int E) {
    int t = blockIdx.x * blockDim.x + threadIdx.x;
    if (t < E) atomicAdd(&deg[dst[t]], 1.0f);
}
__global__ void recip_kernel(float* __restrict__ deg, int n) {
    int t = blockIdx.x * blockDim.x + threadIdx.x;
    if (t < n) deg[t] = 1.0f / fmaxf(deg[t], 1.0f);
}

// ---------------------------------------------------------------------------
// Scatter-add propagation: agg[dst,0:d] += x[src,0:d] * (rowscale ? rowscale[src] : 1)
// thread t -> (edge e, 4-col group vi); accumulators stay L2-resident.
// ---------------------------------------------------------------------------
__global__ void scatter_kernel(const float* __restrict__ x, int xStride,
                               float* __restrict__ agg, int aggStride,
                               const int* __restrict__ src, const int* __restrict__ dst,
                               const float* __restrict__ rowscale,
                               int E, int vecs /* d/4 */) {
    int t = blockIdx.x * blockDim.x + threadIdx.x;
    int total = E * vecs;
    if (t >= total) return;
    int e = t / vecs, vi = t - e * vecs;
    int c = vi * 4;
    int s = src[e];
    float4 v = *(const float4*)(x + (size_t)s * xStride + c);
    if (rowscale) {
        float sc = rowscale[s];
        v.x *= sc; v.y *= sc; v.z *= sc; v.w *= sc;
    }
    float* p = agg + (size_t)dst[e] * aggStride + c;
    atomicAdd(p + 0, v.x);
    atomicAdd(p + 1, v.y);
    atomicAdd(p + 2, v.z);
    atomicAdd(p + 3, v.w);
}

// ---------------------------------------------------------------------------
// WMMA bf16 GEMM: C[n,64] = act(A[n,K]_bf16 @ W[K,64]_bf16 + bias), fp32 accum.
// One wave computes a 16(M) x 64(N) slab = 4 D-tiles of 16x16.
// Software-pipelined ping-pong: chunks c/c+1 compute while c+2/c+3 load.
// All fragment loads use one per-lane base pointer + immediate offsets.
// ---------------------------------------------------------------------------
__global__ __launch_bounds__(256)
void gemm_bf16_kernel(const unsigned short* __restrict__ A,  // bf16 [n, K]
                      const unsigned* __restrict__ Wpk,      // fragment-major packed
                      const float* __restrict__ bias,        // [64]
                      float* __restrict__ C,                 // [n, 64]
                      int n, int K, int relu) {
    const int wave  = threadIdx.x >> 5;
    const int lane  = threadIdx.x & 31;
    const int tileM = (blockIdx.x * 8 + wave) * 16;
    if (tileM >= n) return;                 // wave-uniform exit: EXEC stays all-1 for WMMA

    const int mlan  = lane & 15;
    const int khalf = lane >> 4;
    int m = tileM + mlan;
    if (m >= n) m = n - 1;                  // clamp loads, stores predicated later

    // Per-lane base pointers; all chunk/tile offsets are immediates from here.
    //   A chunk stride = 64B (4 uint4); runs at +0 and +32B (uint4 idx 0,2)
    //   B chunk stride = 4096B (256 uint4); tile stride = 1024B (64 uint4); runs dense
    const uint4* ap = (const uint4*)(A + (size_t)m * K + 8 * khalf);
    const uint4* bp = (const uint4*)(Wpk + (size_t)(mlan * 2 + khalf) * 8);

    const int NC = K >> 5;                  // chunks: 4, 6, 8 or 16 (always even)

    v8f acc[4] = {};
    v8u a0, a1, b0[4], b1[4];

    // prologue: chunks 0 and 1 in flight
    a0 = ld8(ap, 2);
#pragma unroll
    for (int nt = 0; nt < 4; ++nt) b0[nt] = ld8(bp + nt * 64, 1);
    a1 = ld8(ap + 4, 2);
#pragma unroll
    for (int nt = 0; nt < 4; ++nt) b1[nt] = ld8(bp + 256 + nt * 64, 1);

    const uint4* apn = ap + 8;              // chunk c+2 base
    const uint4* bpn = bp + 512;

    for (int c = 0; c + 2 < NC; c += 2) {
        // compute chunk c
        {
            v16bf af = __builtin_bit_cast(v16bf, a0);
#pragma unroll
            for (int nt = 0; nt < 4; ++nt)
                acc[nt] = __builtin_amdgcn_wmma_f32_16x16x32_bf16(
                    false, af, false, __builtin_bit_cast(v16bf, b0[nt]), (short)0,
                    acc[nt], false, false);
        }
        // load chunk c+2
        a0 = ld8(apn, 2);
#pragma unroll
        for (int nt = 0; nt < 4; ++nt) b0[nt] = ld8(bpn + nt * 64, 1);
        // compute chunk c+1
        {
            v16bf af = __builtin_bit_cast(v16bf, a1);
#pragma unroll
            for (int nt = 0; nt < 4; ++nt)
                acc[nt] = __builtin_amdgcn_wmma_f32_16x16x32_bf16(
                    false, af, false, __builtin_bit_cast(v16bf, b1[nt]), (short)0,
                    acc[nt], false, false);
        }
        // load chunk c+3 (always valid: NC even, c <= NC-4)
        a1 = ld8(apn + 4, 2);
#pragma unroll
        for (int nt = 0; nt < 4; ++nt) b1[nt] = ld8(bpn + 256 + nt * 64, 1);

        apn += 8;
        bpn += 512;
    }

    // epilogue: last two chunks, no more loads
    {
        v16bf af = __builtin_bit_cast(v16bf, a0);
#pragma unroll
        for (int nt = 0; nt < 4; ++nt)
            acc[nt] = __builtin_amdgcn_wmma_f32_16x16x32_bf16(
                false, af, false, __builtin_bit_cast(v16bf, b0[nt]), (short)0,
                acc[nt], false, false);
    }
    {
        v16bf af = __builtin_bit_cast(v16bf, a1);
#pragma unroll
        for (int nt = 0; nt < 4; ++nt)
            acc[nt] = __builtin_amdgcn_wmma_f32_16x16x32_bf16(
                false, af, false, __builtin_bit_cast(v16bf, b1[nt]), (short)0,
                acc[nt], false, false);
    }

    // ---- epilogue: bias (+ReLU), predicated fp32 stores ----
    const int mbase = tileM + khalf * 8;
#pragma unroll
    for (int nt = 0; nt < 4; ++nt) {
        int nn = nt * 16 + mlan;
        float bv = bias[nn];
#pragma unroll
        for (int r = 0; r < 8; ++r) {
            int mm = mbase + r;
            if (mm < n) {
                float v = acc[nt][r] + bv;
                if (relu) v = fmaxf(v, 0.0f);
                C[(size_t)mm * 64 + nn] = v;
            }
        }
    }
}

// ---------------------------------------------------------------------------
// Recurrent hop-attention chain; one wave32 per node, 2 columns per lane.
// Emits combined = [att0*o0 | att1*o1 | att2*o2] directly as bf16 [n,192].
// ---------------------------------------------------------------------------
__device__ __forceinline__ float wred32(float v) {
#pragma unroll
    for (int o = 16; o >= 1; o >>= 1) v += __shfl_xor(v, o, 32);
    return v;
}

__global__ __launch_bounds__(256)
void attention_kernel(const float* __restrict__ o0, const float* __restrict__ o1,
                      const float* __restrict__ o2, const float* __restrict__ raw,
                      const float* __restrict__ lrw /*[128]*/, const float* __restrict__ lrb,
                      unsigned short* __restrict__ combined /* bf16 [n,192] */, int n) {
    int wave = threadIdx.x >> 5, lane = threadIdx.x & 31;
    int node = blockIdx.x * 8 + wave;
    if (node >= n) return;
    size_t base = (size_t)node * 64;
    int c0 = lane, c1 = lane + 32;

    float a0 = o0[base + c0],  a1 = o0[base + c1];
    float r0 = raw[base + c0], r1 = raw[base + c1];
    float p0 = o1[base + c0],  p1 = o1[base + c1];
    float q0 = o2[base + c0],  q1 = o2[base + c1];
    float wa0 = lrw[c0], wa1 = lrw[c1], wb0 = lrw[64 + c0], wb1 = lrw[64 + c1];
    float bias = lrb[0];

    float s0 = wred32(a0 * wa0 + a1 * wa1 + r0 * wb0 + r1 * wb1) + bias;
    s0 = 1.0f / (1.0f + __expf(-s0));
    float s1 = wred32(a0 * wa0 + a1 * wa1 + p0 * wb0 + p1 * wb1) + bias;
    s1 = 1.0f / (1.0f + __expf(-s1));
    float m2 = fmaxf(s0, s1);
    float e0 = __expf(s0 - m2), e1 = __expf(s1 - m2);
    float inv2 = 1.0f / (e0 + e1);
    float h0 = e0 * inv2 * a0 + e1 * inv2 * p0;
    float h1 = e0 * inv2 * a1 + e1 * inv2 * p1;
    float s2 = wred32(h0 * wa0 + h1 * wa1 + q0 * wb0 + q1 * wb1) + bias;
    s2 = 1.0f / (1.0f + __expf(-s2));
    float m3 = fmaxf(fmaxf(s0, s1), s2);
    float f0 = __expf(s0 - m3), f1 = __expf(s1 - m3), f2 = __expf(s2 - m3);
    float is = 1.0f / (f0 + f1 + f2);

    size_t cb = (size_t)node * 192;
    combined[cb + c0]        = f2bf(f0 * is * a0);
    combined[cb + c1]        = f2bf(f0 * is * a1);
    combined[cb + 64 + c0]   = f2bf(f1 * is * p0);
    combined[cb + 64 + c1]   = f2bf(f1 * is * p1);
    combined[cb + 128 + c0]  = f2bf(f2 * is * q0);
    combined[cb + 128 + c1]  = f2bf(f2 * is * q1);
}

// ---------------------------------------------------------------------------
// log_softmax over 64 logits; one wave32 per node, 2 cols per lane
// ---------------------------------------------------------------------------
__global__ __launch_bounds__(256)
void logsoftmax_kernel(const float* __restrict__ logits, float* __restrict__ out, int n) {
    int wave = threadIdx.x >> 5, lane = threadIdx.x & 31;
    int node = blockIdx.x * 8 + wave;
    if (node >= n) return;
    size_t base = (size_t)node * 64;
    float v0 = logits[base + lane], v1 = logits[base + lane + 32];
    float m = fmaxf(v0, v1);
#pragma unroll
    for (int o = 16; o >= 1; o >>= 1) m = fmaxf(m, __shfl_xor(m, o, 32));
    float s = __expf(v0 - m) + __expf(v1 - m);
    s = wred32(s);
    float ls = __logf(s);
    out[base + lane]      = v0 - m - ls;
    out[base + lane + 32] = v1 - m - ls;
}

// ---------------------------------------------------------------------------
// Orchestration
// ---------------------------------------------------------------------------
static inline int cdiv(long long a, long long b) { return (int)((a + b - 1) / b); }

extern "C" void kernel_launch(void* const* d_in, const int* in_sizes, int n_in,
                              void* d_out, int out_size, void* d_ws, size_t ws_size,
                              hipStream_t stream) {
    const float* h    = (const float*)d_in[0];
    const int*   esrc = (const int*)d_in[1];
    const int*   edst = (const int*)d_in[2];
    const float* t1w  = (const float*)d_in[3];
    const float* t1b  = (const float*)d_in[4];
    const float* lrw  = (const float*)d_in[5];
    const float* lrb  = (const float*)d_in[6];
    const float* t2w  = (const float*)d_in[7];
    const float* t2b  = (const float*)d_in[8];
    const float* wb0  = (const float*)d_in[9];
    const float* bb0  = (const float*)d_in[10];
    const float* wb1  = (const float*)d_in[11];
    const float* bb1  = (const float*)d_in[12];
    const float* wb2  = (const float*)d_in[13];
    const float* bb2  = (const float*)d_in[14];

    const int IN = 128;
    const int N = in_sizes[0] / IN;
    const int E = in_sizes[1];

    // ---- workspace carve-out (256B aligned) ----
    char* ws = (char*)d_ws;
    size_t off = 0;
    auto carve = [&](size_t bytes) -> char* {
        char* p = ws + off;
        off = (off + bytes + 255) & ~(size_t)255;
        return p;
    };
    float* deg       = (float*)carve((size_t)N * 4);                 // deg -> deg_inv in place
    float* x         = (float*)carve((size_t)N * 64 * 4);            // t1 output (= raw)
    float* concat0   = (float*)carve((size_t)N * 128 * 4);           // unscaled hop sums
    float* concat1   = (float*)carve((size_t)N * 256 * 4);
    float* concat2   = (float*)carve((size_t)N * 512 * 4);
    float* o0        = (float*)carve((size_t)N * 64 * 4);
    float* o1        = (float*)carve((size_t)N * 64 * 4);
    float* o2        = (float*)carve((size_t)N * 64 * 4);
    unsigned short* stage = (unsigned short*)carve((size_t)N * 512 * 2); // bf16 staging / combined
    unsigned* pk_t1  = (unsigned*)carve((size_t)(128 / 32) * 1024 * 4);
    unsigned* pk_b0  = (unsigned*)carve((size_t)(128 / 32) * 1024 * 4);
    unsigned* pk_b1  = (unsigned*)carve((size_t)(256 / 32) * 1024 * 4);
    unsigned* pk_b2  = (unsigned*)carve((size_t)(512 / 32) * 1024 * 4);
    unsigned* pk_t2  = (unsigned*)carve((size_t)(192 / 32) * 1024 * 4);
    float* logits    = concat0;   // reuse (free after block-1 propagation)

    const int T = 256;
    const int gGemm = cdiv(N, 128);       // 8 waves x 16 rows per block
    const int gNode = cdiv(N, 8);         // wave-per-node kernels

    // ---- pack weights to fragment-major bf16 pair-dwords ----
    pack_w_kernel<<<cdiv(4LL * 1024, T), T, 0, stream>>>(t1w, pk_t1, 128);
    pack_w_kernel<<<cdiv(4LL * 1024, T), T, 0, stream>>>(wb0, pk_b0, 128);
    pack_w_kernel<<<cdiv(8LL * 1024, T), T, 0, stream>>>(wb1, pk_b1, 256);
    pack_w_kernel<<<cdiv(16LL * 1024, T), T, 0, stream>>>(wb2, pk_b2, 512);
    pack_w_kernel<<<cdiv(6LL * 1024, T), T, 0, stream>>>(t2w, pk_t2, 192);

    // ---- degrees ----
    hipMemsetAsync(deg, 0, (size_t)N * 4, stream);
    deg_kernel<<<cdiv(E, T), T, 0, stream>>>(edst, deg, E);
    recip_kernel<<<cdiv(N, T), T, 0, stream>>>(deg, N);

    // ---- t1: x = relu(h @ t1_w + b) ----
    cvt_bf16_kernel<<<cdiv((long long)N * IN, T), T, 0, stream>>>(h, nullptr, 1, stage, (size_t)N * IN);
    gemm_bf16_kernel<<<gGemm, T, 0, stream>>>(stage, pk_t1, t1b, x, N, 128, 1);

    // one propagation hop with lazy scaling of the SOURCE rows
    auto hop = [&](const float* src, int srcStride, float* dst, int dstStride, int d,
                   const float* srcScale) {
        long long total = (long long)E * (d / 4);
        scatter_kernel<<<cdiv(total, T), T, 0, stream>>>(src, srcStride, dst, dstStride,
                                                         esrc, edst, srcScale, E, d / 4);
    };
    // convert a concat buffer (unscaled sums) to bf16 with deg_inv row scaling
    auto cvt_scaled = [&](const float* src, int d) {
        cvt_bf16_kernel<<<cdiv((long long)N * d, T), T, 0, stream>>>(src, deg, d, stage,
                                                                     (size_t)N * d);
    };

    // ---- block 0: two hops on d=64, concat -> [N,128], o0 = relu(. @ wb0) ----
    hipMemsetAsync(concat0, 0, (size_t)N * 128 * 4, stream);
    hop(x, 64, concat0, 128, 64, nullptr);            // hop1 = sum(x[src])
    hop(concat0, 128, concat0 + 64, 128, 64, deg);    // hop2 = sum(deg_inv*hop1[src])
    cvt_scaled(concat0, 128);
    gemm_bf16_kernel<<<gGemm, T, 0, stream>>>(stage, pk_b0, bb0, o0, N, 128, 1);

    // ---- block 1: two hops on d=128, concat -> [N,256], o1 ----
    hipMemsetAsync(concat1, 0, (size_t)N * 256 * 4, stream);
    hop(concat0, 128, concat1, 256, 128, deg);
    hop(concat1, 256, concat1 + 128, 256, 128, deg);
    cvt_scaled(concat1, 256);
    gemm_bf16_kernel<<<gGemm, T, 0, stream>>>(stage, pk_b1, bb1, o1, N, 256, 1);

    // ---- block 2: two hops on d=256, concat -> [N,512], o2 ----
    hipMemsetAsync(concat2, 0, (size_t)N * 512 * 4, stream);
    hop(concat1, 256, concat2, 512, 256, deg);
    hop(concat2, 512, concat2 + 256, 512, 256, deg);
    cvt_scaled(concat2, 512);
    gemm_bf16_kernel<<<gGemm, T, 0, stream>>>(stage, pk_b2, bb2, o2, N, 512, 1);

    // ---- attention chain -> combined bf16 [N,192] (reuse stage) ----
    attention_kernel<<<gNode, T, 0, stream>>>(o0, o1, o2, x, lrw, lrb, stage, N);

    // ---- logits = combined @ t2_w + b ; log_softmax -> d_out ----
    gemm_bf16_kernel<<<gGemm, T, 0, stream>>>(stage, pk_t2, t2b, logits, N, 192, 0);
    logsoftmax_kernel<<<gNode, T, 0, stream>>>(logits, (float*)d_out, N);
}
